// EncoderModel_81844896792972
// MI455X (gfx1250) — compile-verified, matrix operand order checked by reading, not run on hardware
//
#include <hip/hip_runtime.h>
#include <hip/hip_bf16.h>
#include <math.h>

// ---------------------------------------------------------------------------
// Types for CDNA5 WMMA intrinsics (gfx1250, wave32)
// ---------------------------------------------------------------------------
typedef __attribute__((ext_vector_type(16))) __bf16 bf16x16;
typedef __attribute__((ext_vector_type(8)))  __bf16 bf16x8;
typedef __attribute__((ext_vector_type(8)))  float  f32x8;

#define PAD_IDX 0
#define NEG_INF (-__builtin_inff())

__device__ __forceinline__ f32x8 wmma_bf16(bf16x16 a, bf16x16 b, f32x8 c) {
    // D = A(16x32 bf16) x B(32x16 bf16) + C(16x16 f32)
    return __builtin_amdgcn_wmma_f32_16x16x32_bf16(
        false, a, false, b, (short)0, c, false, false);
}

// ---------------------------------------------------------------------------
// Unified fragment loader (ISA 7.12.2, wave32).
//   A-fragment from a row-major 16xK tile (row = M index per lane)
//   B-fragment from an n-major tile (row = N index per lane, value(k,n)=p[n*ld+k])
// Per lane: two contiguous 16-byte runs -> 2x (ds|global)_load_b128.
// ---------------------------------------------------------------------------
__device__ __forceinline__ bf16x16 load_frag(const __bf16* p, int ld, int lane) {
    bf16x16 f;
    const int row = lane & 15;
    const int h8  = (lane >> 4) << 3;     // 0 or 8
    const __bf16* r = p + (size_t)row * ld + h8;
#pragma unroll
    for (int e = 0; e < 8; ++e) {         // K = h8..h8+7
        const int vr = e >> 1;            // VGPR 0..3, K = 2*vr + (e&1)
        f[2 * vr + (e & 1)] = r[e];
    }
#pragma unroll
    for (int e = 0; e < 8; ++e) {         // K = h8+16..h8+23
        const int vr = 4 + (e >> 1);      // VGPR 4..7
        f[2 * vr + (e & 1)] = r[16 + e];
    }
    return f;
}

// LDS byte offset of a shared-memory pointer (flat LDS addr low 32 bits)
__device__ __forceinline__ uint32_t lds_off(const void* p) {
    return (uint32_t)(uintptr_t)p;
}

// CDNA5 async global->LDS 128-bit copy (per-lane), tracked by ASYNCcnt.
__device__ __forceinline__ void async_copy_b128(uint32_t lds_byte_off,
                                                const void* gaddr) {
    asm volatile("global_load_async_to_lds_b128 %0, %1, off"
                 :: "v"(lds_byte_off), "v"(gaddr) : "memory");
}
// Wait until at most `n` async ops outstanding (in-order completion).
__device__ __forceinline__ void wait_async0() {
    asm volatile("s_wait_asynccnt 0x0" ::: "memory");
}
__device__ __forceinline__ void wait_async4() {
    asm volatile("s_wait_asynccnt 0x4" ::: "memory");
}

// ---------------------------------------------------------------------------
// f32 (L,K,N) -> bf16 transposed (L,N,K): Wt[l][n][k] = W[l][k][n]
// ---------------------------------------------------------------------------
__global__ __launch_bounds__(256) void cvt_f32_bf16_T(const float* __restrict__ W,
                                                      __bf16* __restrict__ Wt,
                                                      int K, int N, long total) {
    long i = (long)blockIdx.x * blockDim.x + threadIdx.x;
    long stride = (long)gridDim.x * blockDim.x;
    const long per = (long)K * N;
    for (; i < total; i += stride) {
        const long l = i / per, rem = i - l * per;
        const long n = rem / K, k = rem - n * K;
        Wt[i] = (__bf16)W[l * per + k * N + n];
    }
}

// ---------------------------------------------------------------------------
// Embedding gather + positional encoding
// ---------------------------------------------------------------------------
__global__ __launch_bounds__(128) void embed_kernel(const int* __restrict__ x,
                                                    const float* __restrict__ embed,
                                                    const float* __restrict__ pe,
                                                    float* __restrict__ hf,
                                                    __bf16* __restrict__ hb,
                                                    int T, int D) {
    const int bt  = blockIdx.x;
    const int tok = x[bt];
    const int t   = bt % T;
    for (int dd = threadIdx.x; dd < D; dd += 128) {
        float v = embed[(size_t)tok * D + dd] + pe[(size_t)t * D + dd];
        hf[(size_t)bt * D + dd] = v;
        hb[(size_t)bt * D + dd] = (__bf16)v;
    }
}

// ---------------------------------------------------------------------------
// Tiled WMMA GEMM:  C[MxN] = A[MxK](bf16) * B[KxN] + bias
//   B pre-transposed (Bt[N][K], n-major) -> all fragment reads k-contiguous.
//   Block tile 128x128, 8 waves -> wave tile 32x64 (2x4 WMMA accumulators).
//   Double-buffered LDS staged with global_load_async_to_lds_b128:
//   issue tile i+1, s_wait_asynccnt 4 -> tile i landed, tile i+1 in flight.
//   outT=1 writes the bf16 result transposed: Cb[n*M + m] (for V^T).
// ---------------------------------------------------------------------------
#define GBM 128
#define GBN 128
#define GKB 32

__global__ __launch_bounds__(256) void gemm_bf16_wmma(
    const __bf16* __restrict__ A, const __bf16* __restrict__ Bt,
    const float* __restrict__ bias,
    float* __restrict__ Cf, __bf16* __restrict__ Cb,
    int M, int N, int K, int relu, int outT) {
    __shared__ __bf16 As[2][GBM][GKB];   // 2 x 8 KB, m-major
    __shared__ __bf16 Bs[2][GBN][GKB];   // 2 x 8 KB, n-major

    const int tid  = threadIdx.x;
    const int lane = tid & 31;
    const int wave = tid >> 5;
    const int wm   = wave >> 1;          // 0..3 -> 32 rows each
    const int wn   = wave & 1;           // 0..1 -> 64 cols each
    const int bm   = blockIdx.y * GBM;
    const int bn   = blockIdx.x * GBN;

    // 512 chunks of 8 bf16 per tile, 2 per thread (rows of 32 = 4 chunks)
    const int r0 = tid >> 2,         c0 = (tid & 3) * 8;
    const int r1 = (tid + 256) >> 2, c1 = ((tid + 256) & 3) * 8;

    auto issue_tile = [&](int buf, int k0) {
        async_copy_b128(lds_off(&As[buf][r0][c0]),
                        A  + (size_t)(bm + r0) * K + k0 + c0);
        async_copy_b128(lds_off(&As[buf][r1][c1]),
                        A  + (size_t)(bm + r1) * K + k0 + c1);
        async_copy_b128(lds_off(&Bs[buf][r0][c0]),
                        Bt + (size_t)(bn + r0) * K + k0 + c0);
        async_copy_b128(lds_off(&Bs[buf][r1][c1]),
                        Bt + (size_t)(bn + r1) * K + k0 + c1);
        if (k0 + GKB < K) {  // L2 prefetch of the slab after next
            __builtin_prefetch(A  + (size_t)(bm + r0) * K + k0 + GKB + c0, 0, 1);
            __builtin_prefetch(Bt + (size_t)(bn + r0) * K + k0 + GKB + c0, 0, 1);
        }
    };

    f32x8 acc[2][4] = {};
    const int nk = K / GKB;

    issue_tile(0, 0);
    for (int i = 0; i < nk; ++i) {
        const int buf = i & 1;
        const bool more = (i + 1 < nk);
        if (more) issue_tile(buf ^ 1, (i + 1) * GKB);
        if (more) wait_async4();   // tile i landed; tile i+1 still in flight
        else      wait_async0();
        __syncthreads();

        bf16x16 a0 = load_frag(&As[buf][wm * 32][0],      GKB, lane);
        bf16x16 a1 = load_frag(&As[buf][wm * 32 + 16][0], GKB, lane);
        bf16x16 b[4];
#pragma unroll
        for (int ni = 0; ni < 4; ++ni)
            b[ni] = load_frag(&Bs[buf][wn * 64 + ni * 16][0], GKB, lane);

#pragma unroll
        for (int ni = 0; ni < 4; ++ni) {
            acc[0][ni] = wmma_bf16(a0, b[ni], acc[0][ni]);
            acc[1][ni] = wmma_bf16(a1, b[ni], acc[1][ni]);
        }
        __syncthreads();   // all waves done reading buf before it is re-filled
    }

    // Epilogue: C/D layout -> lane holds col (lane&15), rows r + 8*(lane>>4)
    const int ncol = lane & 15;
    const int rofs = (lane >> 4) << 3;
#pragma unroll
    for (int mi = 0; mi < 2; ++mi) {
#pragma unroll
        for (int ni = 0; ni < 4; ++ni) {
            const int gcol = bn + wn * 64 + ni * 16 + ncol;
            const float bv = bias ? bias[gcol] : 0.f;
            if (outT && Cb) {
                // 8 contiguous rows along M -> single 16B store per lane
                bf16x8 v8;
#pragma unroll
                for (int r = 0; r < 8; ++r) {
                    float v = acc[mi][ni][r] + bv;
                    if (relu) v = fmaxf(v, 0.f);
                    v8[r] = (__bf16)v;
                }
                const int growb = bm + wm * 32 + mi * 16 + rofs;
                *(bf16x8*)&Cb[(size_t)gcol * M + growb] = v8;
            } else {
#pragma unroll
                for (int r = 0; r < 8; ++r) {
                    const int grow = bm + wm * 32 + mi * 16 + r + rofs;
                    float v = acc[mi][ni][r] + bv;
                    if (relu) v = fmaxf(v, 0.f);
                    if (Cf) Cf[(size_t)grow * N + gcol] = v;
                    if (Cb) Cb[(size_t)grow * N + gcol] = (__bf16)v;
                }
            }
        }
    }
}

// ---------------------------------------------------------------------------
// Flash attention: one wave per (b, head, 16-row q tile). dk = 64.
//   S = Q K^T via WMMA (K=64, chained 16x16x32); online softmax in LDS;
//   P (bf16, LDS) x V^T via WMMA (V supplied transposed: vT[D][M]).
// ---------------------------------------------------------------------------
__global__ __launch_bounds__(32) void attn_kernel(
    const __bf16* __restrict__ qb, const __bf16* __restrict__ kb,
    const __bf16* __restrict__ vT, const int* __restrict__ x,
    __bf16* __restrict__ ctx, int Bn, int T, int D, int Hh, int Mrows) {
    const int nqt = T / 16;
    const int qt = blockIdx.x % nqt;
    const int h  = (blockIdx.x / nqt) % Hh;
    const int b  = blockIdx.x / (nqt * Hh);
    const int lane = threadIdx.x;
    const int dk = D / Hh;                      // 64
    const size_t bT = (size_t)b * T;
    const size_t headoff = (size_t)h * dk;

    __shared__ float  sM[16], sL[16], sFac[16];
    __shared__ float  sS[16][32];
    __shared__ __bf16 sP[16][32];

    if (lane < 16) { sM[lane] = NEG_INF; sL[lane] = 0.f; }
    __syncthreads();

    // Q fragments for K-dim = 64 (two 32-chunks)
    bf16x16 aq[2];
#pragma unroll
    for (int c = 0; c < 2; ++c)
        aq[c] = load_frag(qb + (bT + qt * 16) * D + headoff + c * 32, D, lane);

    f32x8 of[4] = {};
    const float scale = 0.125f;                 // 1/sqrt(64)
    const int ncol = lane & 15;
    const int rofs = (lane >> 4) << 3;

    for (int kt = 0; kt < T / 32; ++kt) {
        // ---- S = Q * K^T for 32 keys (2 groups of 16 columns) ----
#pragma unroll
        for (int g = 0; g < 2; ++g) {
            // K rows are n-major in memory ([token][D]) -> load_frag fits
            const __bf16* kp = kb + (bT + kt * 32 + g * 16) * D + headoff;
            f32x8 s = {};
            s = wmma_bf16(aq[0], load_frag(kp,      D, lane), s);
            s = wmma_bf16(aq[1], load_frag(kp + 32, D, lane), s);
            const int key = kt * 32 + g * 16 + ncol;
            const bool pad = (x[bT + key] == PAD_IDX);
#pragma unroll
            for (int r = 0; r < 8; ++r)
                sS[r + rofs][g * 16 + ncol] = pad ? NEG_INF : s[r] * scale;
        }
        __syncthreads();

        // ---- online softmax (one lane per row) ----
        if (lane < 16) {
            const float oldm = sM[lane];
            float mx = oldm;
#pragma unroll
            for (int j = 0; j < 32; ++j) mx = fmaxf(mx, sS[lane][j]);
            const float fac = (oldm == NEG_INF) ? 0.f : __expf(oldm - mx);
            float l = sL[lane] * fac;
#pragma unroll
            for (int j = 0; j < 32; ++j) {
                const float pv = (mx == NEG_INF) ? 0.f : __expf(sS[lane][j] - mx);
                sP[lane][j] = (__bf16)pv;
                l += pv;
            }
            sM[lane] = mx; sL[lane] = l; sFac[lane] = fac;
        }
        __syncthreads();

        // ---- rescale running output, then O += P * V ----
        bf16x16 ap = load_frag(&sP[0][0], 32, lane);
#pragma unroll
        for (int g = 0; g < 4; ++g) {
#pragma unroll
            for (int r = 0; r < 8; ++r) of[g][r] *= sFac[r + rofs];
            // V^T: value(k=token, n=dk col) = vT[(headoff+n)*M + token]
            const __bf16* vp = vT + (headoff + g * 16) * (size_t)Mrows
                                  + bT + kt * 32;
            of[g] = wmma_bf16(ap, load_frag(vp, Mrows, lane), of[g]);
        }
        __syncthreads();
    }

    // ---- normalize and emit context (bf16, GEMM-ready) ----
#pragma unroll
    for (int g = 0; g < 4; ++g) {
#pragma unroll
        for (int r = 0; r < 8; ++r) {
            const int row = r + rofs;
            const float l = sL[row];
            const float v = (l > 0.f) ? of[g][r] / l : 0.f;
            ctx[(bT + qt * 16 + row) * D + headoff + g * 16 + ncol] = (__bf16)v;
        }
    }
}

// ---------------------------------------------------------------------------
// Residual + LayerNorm (D = 512): h = LN(h + sub) ; writes f32 + bf16 shadow
// ---------------------------------------------------------------------------
__global__ __launch_bounds__(256) void ln_kernel(
    float* __restrict__ hf, const float* __restrict__ sub,
    const float* __restrict__ g, const float* __restrict__ bb,
    __bf16* __restrict__ hb, int D) {
    const int tid = threadIdx.x;
    const size_t base = (size_t)blockIdx.x * D;
    __shared__ float red[256];

    float v0 = hf[base + tid]       + sub[base + tid];
    float v1 = hf[base + tid + 256] + sub[base + tid + 256];

    red[tid] = v0 + v1;
    __syncthreads();
    for (int s = 128; s > 0; s >>= 1) {
        if (tid < s) red[tid] += red[tid + s];
        __syncthreads();
    }
    const float mean = red[0] * (1.f / 512.f);
    __syncthreads();

    const float d0 = v0 - mean, d1 = v1 - mean;
    red[tid] = d0 * d0 + d1 * d1;
    __syncthreads();
    for (int s = 128; s > 0; s >>= 1) {
        if (tid < s) red[tid] += red[tid + s];
        __syncthreads();
    }
    const float inv = rsqrtf(red[0] * (1.f / 512.f) + 1e-5f);

    const float o0 = d0 * inv * g[tid]       + bb[tid];
    const float o1 = d1 * inv * g[tid + 256] + bb[tid + 256];
    hf[base + tid]       = o0;  hb[base + tid]       = (__bf16)o0;
    hf[base + tid + 256] = o1;  hb[base + tid + 256] = (__bf16)o1;
}

// ---------------------------------------------------------------------------
// Output head: sent = cls@Ws+bs (8x3), len = cls@Wl+bl (8x3), cls copy (8x512)
// ---------------------------------------------------------------------------
__global__ __launch_bounds__(64) void head_kernel(
    const float* __restrict__ hf,
    const float* __restrict__ sW, const float* __restrict__ sb,
    const float* __restrict__ lW, const float* __restrict__ lb,
    float* __restrict__ out, int Bn, int T, int D) {
    const int b = blockIdx.x;
    const int tid = threadIdx.x;
    const float* cls = hf + (size_t)b * T * D;   // token 0 of sequence b
    if (tid < 3) {
        float s = sb[tid];
        for (int d = 0; d < D; ++d) s += cls[d] * sW[(size_t)d * 3 + tid];
        out[b * 3 + tid] = s;
    } else if (tid < 6) {
        const int j = tid - 3;
        float s = lb[j];
        for (int d = 0; d < D; ++d) s += cls[d] * lW[(size_t)d * 3 + j];
        out[Bn * 3 + b * 3 + j] = s;
    }
    for (int d = tid; d < D; d += 64)
        out[Bn * 6 + (size_t)b * D + d] = cls[d];
}

// ---------------------------------------------------------------------------
// Host orchestration
// ---------------------------------------------------------------------------
extern "C" void kernel_launch(void* const* d_in, const int* in_sizes, int n_in,
                              void* d_out, int out_size, void* d_ws, size_t ws_size,
                              hipStream_t stream) {
    (void)in_sizes; (void)n_in; (void)out_size; (void)ws_size;
    const int*   x     = (const int*)  d_in[0];
    const float* embed = (const float*)d_in[1];
    const float* pe    = (const float*)d_in[2];
    const float* Wq    = (const float*)d_in[3];
    const float* bq    = (const float*)d_in[4];
    const float* Wk    = (const float*)d_in[5];
    const float* bk    = (const float*)d_in[6];
    const float* Wv    = (const float*)d_in[7];
    const float* bv    = (const float*)d_in[8];
    const float* Wo    = (const float*)d_in[9];
    const float* bo    = (const float*)d_in[10];
    const float* ln1g  = (const float*)d_in[11];
    const float* ln1b  = (const float*)d_in[12];
    const float* W1    = (const float*)d_in[13];
    const float* b1    = (const float*)d_in[14];
    const float* W2    = (const float*)d_in[15];
    const float* b2    = (const float*)d_in[16];
    const float* ln2g  = (const float*)d_in[17];
    const float* ln2b  = (const float*)d_in[18];
    const float* sW    = (const float*)d_in[19];
    const float* sb    = (const float*)d_in[20];
    const float* lW    = (const float*)d_in[21];
    const float* lb    = (const float*)d_in[22];

    const int Bn = 8, T = 2048, D = 512, F = 2048, L = 6, Hh = 8;
    const size_t M = (size_t)Bn * T;               // 16384 rows

    // --- workspace carve-out (256B aligned bumps) ---
    char* p = (char*)d_ws;
    auto bump = [&](size_t bytes) -> void* {
        void* r = (void*)p;
        p += (bytes + 255) & ~(size_t)255;
        return r;
    };
    float*  hf   = (float*) bump(M * D * sizeof(float));
    float*  t0f  = (float*) bump(M * D * sizeof(float));
    __bf16* hb   = (__bf16*)bump(M * D * sizeof(__bf16));
    __bf16* qb   = (__bf16*)bump(M * D * sizeof(__bf16));
    __bf16* kbuf = (__bf16*)bump(M * D * sizeof(__bf16));
    __bf16* vT   = (__bf16*)bump(M * D * sizeof(__bf16));   // transposed [D][M]
    __bf16* ctxb = (__bf16*)bump(M * D * sizeof(__bf16));
    __bf16* midb = (__bf16*)bump(M * F * sizeof(__bf16));
    __bf16* WqT  = (__bf16*)bump((size_t)L * D * D * sizeof(__bf16));
    __bf16* WkT  = (__bf16*)bump((size_t)L * D * D * sizeof(__bf16));
    __bf16* WvT  = (__bf16*)bump((size_t)L * D * D * sizeof(__bf16));
    __bf16* WoT  = (__bf16*)bump((size_t)L * D * D * sizeof(__bf16));
    __bf16* W1T  = (__bf16*)bump((size_t)L * D * F * sizeof(__bf16));
    __bf16* W2T  = (__bf16*)bump((size_t)L * F * D * sizeof(__bf16));

    // --- weights -> bf16, pre-transposed to [N][K] per layer ---
    const long nqk = (long)L * D * D, nff = (long)L * D * F;
    cvt_f32_bf16_T<<<2048, 256, 0, stream>>>(Wq, WqT, D, D, nqk);
    cvt_f32_bf16_T<<<2048, 256, 0, stream>>>(Wk, WkT, D, D, nqk);
    cvt_f32_bf16_T<<<2048, 256, 0, stream>>>(Wv, WvT, D, D, nqk);
    cvt_f32_bf16_T<<<2048, 256, 0, stream>>>(Wo, WoT, D, D, nqk);
    cvt_f32_bf16_T<<<4096, 256, 0, stream>>>(W1, W1T, D, F, nff);   // Kc=D, Nc=F
    cvt_f32_bf16_T<<<4096, 256, 0, stream>>>(W2, W2T, F, D, nff);   // Kc=F, Nc=D

    // --- embedding + positional encoding ---
    embed_kernel<<<(int)M, 128, 0, stream>>>(x, embed, pe, hf, hb, T, D);

    const dim3 gD(D / GBN, (int)(M / GBM));        // N=512 GEMMs
    const dim3 gF(F / GBN, (int)(M / GBM));        // N=2048 GEMM
    const int attnBlocks = Bn * Hh * (T / 16);

    for (int l = 0; l < L; ++l) {
        const __bf16* wq = WqT + (size_t)l * D * D;
        const __bf16* wk = WkT + (size_t)l * D * D;
        const __bf16* wv = WvT + (size_t)l * D * D;
        const __bf16* wo = WoT + (size_t)l * D * D;
        const __bf16* w1 = W1T + (size_t)l * D * F;
        const __bf16* w2 = W2T + (size_t)l * F * D;

        // Q/K normal layout; V written transposed [D][M] for attention
        gemm_bf16_wmma<<<gD, 256, 0, stream>>>(hb, wq, bq + (size_t)l * D,
                                               nullptr, qb,   (int)M, D, D, 0, 0);
        gemm_bf16_wmma<<<gD, 256, 0, stream>>>(hb, wk, bk + (size_t)l * D,
                                               nullptr, kbuf, (int)M, D, D, 0, 0);
        gemm_bf16_wmma<<<gD, 256, 0, stream>>>(hb, wv, bv + (size_t)l * D,
                                               nullptr, vT,   (int)M, D, D, 0, 1);

        // Multi-head flash attention
        attn_kernel<<<attnBlocks, 32, 0, stream>>>(qb, kbuf, vT, x, ctxb,
                                                   Bn, T, D, Hh, (int)M);

        // Output projection, then residual + LN1
        gemm_bf16_wmma<<<gD, 256, 0, stream>>>(ctxb, wo, bo + (size_t)l * D,
                                               t0f, nullptr, (int)M, D, D, 0, 0);
        ln_kernel<<<(int)M, 256, 0, stream>>>(hf, t0f, ln1g + (size_t)l * D,
                                              ln1b + (size_t)l * D, hb, D);

        // FFN: relu(h@W1+b1) -> bf16 mid; mid@W2+b2 -> f32; residual + LN2
        gemm_bf16_wmma<<<gF, 256, 0, stream>>>(hb, w1, b1 + (size_t)l * F,
                                               nullptr, midb, (int)M, F, D, 1, 0);
        gemm_bf16_wmma<<<gD, 256, 0, stream>>>(midb, w2, b2 + (size_t)l * D,
                                               t0f, nullptr, (int)M, D, F, 0, 0);
        ln_kernel<<<(int)M, 256, 0, stream>>>(hf, t0f, ln2g + (size_t)l * D,
                                              ln2b + (size_t)l * D, hb, D);
    }

    // --- classification heads + cls copy ---
    head_kernel<<<Bn, 64, 0, stream>>>(hf, sW, sb, lW, lb, (float*)d_out,
                                       Bn, T, D);
}
